// MultiHeadLatentAttention_72009421685113
// MI455X (gfx1250) — matrix operand order repaired and checked by reading, not compiled, not run on hardware
//
#include <hip/hip_runtime.h>
#include <math.h>

// ---------------------------------------------------------------------------
// Problem constants (from reference)
// ---------------------------------------------------------------------------
#define BATCH 4
#define SEQ   2048
#define EMB   1024
#define HEADS 16
#define HDIM  64
#define WIN   7
#define M1    (BATCH * SEQ)          // 8192 rows for qkv/proj GEMMs
#define N1    (3 * EMB)              // 3072
#define M2    (BATCH * HEADS * SEQ)  // 131072 rows for lat GEMMs

typedef __attribute__((ext_vector_type(16))) __bf16 v16bf;
typedef __attribute__((ext_vector_type(8)))  __bf16 v8bf;
typedef __attribute__((ext_vector_type(8)))  float  v8f;

// ---------------------------------------------------------------------------
// fp32 -> bf16 (round-to-nearest-even), stored as raw u16
// ---------------------------------------------------------------------------
__device__ inline unsigned short f2bf(float f) {
    union { float f; unsigned int u; } x;
    x.f = f;
    unsigned int u = x.u;
    unsigned int r = u + 0x7FFFu + ((u >> 16) & 1u);
    return (unsigned short)(r >> 16);
}

__global__ __launch_bounds__(256) void cvt_f32_bf16(const float* __restrict__ in,
                                                    unsigned short* __restrict__ out,
                                                    int n) {
    int i = (blockIdx.x * 256 + threadIdx.x) * 4;
    if (i < n) {
        float4 f = *(const float4*)(in + i);
        ushort4 o;
        o.x = f2bf(f.x); o.y = f2bf(f.y); o.z = f2bf(f.z); o.w = f2bf(f.w);
        *(ushort4*)(out + i) = o;
    }
}

// ---------------------------------------------------------------------------
// WMMA fragment loads (ISA 7.12.2, 16-bit 16x32 A / 32x16 B, wave32)
// A lane L (L<16): halves 0-7 = K k0..k0+7,  halves 8-15 = K k0+16..k0+23
//   lane L>=16  : halves 0-7 = K k0+8..+15,  halves 8-15 = K k0+24..+31
// B lane L: col = L%16; same K layout as A
// Both A (M x K) and W (N x K) are row-major over K -> identical access form.
// ---------------------------------------------------------------------------
__device__ inline v16bf load_a_frag(const unsigned short* __restrict__ A, int lda,
                                    int row0, int k0, int lane) {
    int m = row0 + (lane & 15);
    const unsigned short* p = A + (size_t)m * lda + k0 + ((lane & 16) ? 8 : 0);
    v8bf lo = *reinterpret_cast<const v8bf*>(p);
    v8bf hi = *reinterpret_cast<const v8bf*>(p + 16);
    return __builtin_shufflevector(lo, hi, 0,1,2,3,4,5,6,7,8,9,10,11,12,13,14,15);
}

__device__ inline v16bf load_b_frag(const unsigned short* __restrict__ W, int ldb,
                                    int col0, int k0, int lane) {
    int n = col0 + (lane & 15);
    const unsigned short* p = W + (size_t)n * ldb + k0 + ((lane & 16) ? 8 : 0);
    v8bf lo = *reinterpret_cast<const v8bf*>(p);
    v8bf hi = *reinterpret_cast<const v8bf*>(p + 16);
    return __builtin_shufflevector(lo, hi, 0,1,2,3,4,5,6,7,8,9,10,11,12,13,14,15);
}

// ---------------------------------------------------------------------------
// GEMM mainloop: C(32 x NSUB*16) += A(32 x K) * W(NSUB*16 x K)^T
// ---------------------------------------------------------------------------
template<int NSUB>
__device__ inline void gemm_mainloop(const unsigned short* __restrict__ A, int lda,
                                     const unsigned short* __restrict__ W, int ldb,
                                     int rowBase, int colBase, int K, int lane,
                                     v8f (&c)[2][NSUB]) {
    for (int k0 = 0; k0 < K; k0 += 32) {
        v16bf a0 = load_a_frag(A, lda, rowBase,      k0, lane);
        v16bf a1 = load_a_frag(A, lda, rowBase + 16, k0, lane);
        v16bf b[NSUB];
#pragma unroll
        for (int ni = 0; ni < NSUB; ++ni)
            b[ni] = load_b_frag(W, ldb, colBase + ni * 16, k0, lane);
#pragma unroll
        for (int ni = 0; ni < NSUB; ++ni) {
            c[0][ni] = __builtin_amdgcn_wmma_f32_16x16x32_bf16(
                false, a0, false, b[ni], (short)0, c[0][ni], false, false);
            c[1][ni] = __builtin_amdgcn_wmma_f32_16x16x32_bf16(
                false, a1, false, b[ni], (short)0, c[1][ni], false, false);
        }
    }
}

// ---------------------------------------------------------------------------
// QKV GEMM: qkv[M1,3E] = x_bf @ Wqkv_bf^T + b_qkv
// epilogue scatters: q -> f32 [B,H,S,D]; k,v -> bf16 [B,H,S,D]
// block tile 128(M) x 128(N), 8 waves (4M x 2N), wave tile 32x64
// ---------------------------------------------------------------------------
__global__ __launch_bounds__(256) void gemm_qkv(const unsigned short* __restrict__ A,
                                                const unsigned short* __restrict__ W,
                                                const float* __restrict__ bias,
                                                float* __restrict__ q,
                                                unsigned short* __restrict__ kbf,
                                                unsigned short* __restrict__ vbf) {
    int lane = threadIdx.x & 31, wid = threadIdx.x >> 5;
    int rowBase = blockIdx.y * 128 + (wid & 3) * 32;
    int colBase = blockIdx.x * 128 + (wid >> 2) * 64;
    v8f c[2][4] = {};
    gemm_mainloop<4>(A, EMB, W, EMB, rowBase, colBase, EMB, lane, c);

    int cl = lane & 15, rl = (lane & 16) ? 8 : 0;
#pragma unroll
    for (int mi = 0; mi < 2; ++mi)
#pragma unroll
        for (int ni = 0; ni < 4; ++ni) {
            int n = colBase + ni * 16 + cl;
            float bv = bias[n];
            int t  = n >> 10;        // 0=q 1=k 2=v
            int hd = n & 1023;       // h*64+d
            int h = hd >> 6, d = hd & 63;
#pragma unroll
            for (int v = 0; v < 8; ++v) {
                int m = rowBase + mi * 16 + rl + v;
                float val = c[mi][ni][v] + bv;
                int b = m >> 11, s = m & 2047;
                size_t idx = (((size_t)(b * HEADS + h)) * SEQ + s) * HDIM + d;
                if (t == 0)      q[idx]   = val;
                else if (t == 1) kbf[idx] = f2bf(val);
                else             vbf[idx] = f2bf(val);
            }
        }
}

// ---------------------------------------------------------------------------
// lat GEMM: out[M2,64] = in_bf[M2,64] @ Wlat_bf^T + b_lat  (K = 64)
// block tile 128(M) x 64(N), 8 waves (4M x 2N), wave tile 32x32
// ---------------------------------------------------------------------------
__global__ __launch_bounds__(256) void gemm_lat(const unsigned short* __restrict__ A,
                                                const unsigned short* __restrict__ W,
                                                const float* __restrict__ bias,
                                                float* __restrict__ out) {
    int lane = threadIdx.x & 31, wid = threadIdx.x >> 5;
    int rowBase = blockIdx.y * 128 + (wid & 3) * 32;
    int colBase = (wid >> 2) * 32;
    v8f c[2][2] = {};
    gemm_mainloop<2>(A, HDIM, W, HDIM, rowBase, colBase, HDIM, lane, c);

    int cl = lane & 15, rl = (lane & 16) ? 8 : 0;
#pragma unroll
    for (int mi = 0; mi < 2; ++mi)
#pragma unroll
        for (int ni = 0; ni < 2; ++ni) {
            int n = colBase + ni * 16 + cl;
            float bv = bias[n];
#pragma unroll
            for (int v = 0; v < 8; ++v) {
                int m = rowBase + mi * 16 + rl + v;
                out[(size_t)m * HDIM + n] = c[mi][ni][v] + bv;
            }
        }
}

// ---------------------------------------------------------------------------
// Windowed attention: one lane per (b,h,s). Exact reference window padding:
//  s<3          : [0..s+3] then (3-s) copies of S-1 appended
//  3<=s<=S-4    : [s-3 .. s+3]
//  s>S-4        : (s-(S-4)) zeros prepended, then [s-3 .. S-1]
// Output written bf16 in [B,S,E] layout (A matrix of proj GEMM).
// ---------------------------------------------------------------------------
__global__ __launch_bounds__(256) void attn_win(const float* __restrict__ q,
                                                const float* __restrict__ klat,
                                                const float* __restrict__ vlat,
                                                unsigned short* __restrict__ outbf) {
    int t = blockIdx.x * 256 + threadIdx.x;     // 0 .. M2-1
    int s  = t & (SEQ - 1);
    int bh = t >> 11;                            // b*H + h
    int h  = bh & (HEADS - 1);
    int b  = bh >> 4;

    const float4* q4 = (const float4*)(q + (size_t)t * HDIM);
    float4 qr[16];
#pragma unroll
    for (int c = 0; c < 16; ++c) qr[c] = q4[c];

    int w[WIN];
    if (s >= 3 && s <= SEQ - 4) {
#pragma unroll
        for (int j = 0; j < WIN; ++j) w[j] = s - 3 + j;
    } else if (s < 3) {
        int len = s + 4;
#pragma unroll
        for (int j = 0; j < WIN; ++j) w[j] = (j < len) ? j : (SEQ - 1);
    } else {
        int pad = s - (SEQ - 4);
#pragma unroll
        for (int j = 0; j < WIN; ++j) w[j] = (j < pad) ? 0 : (s - 3 + j - pad);
    }

    size_t rowb = (size_t)bh * SEQ;
    float sc[WIN];
#pragma unroll
    for (int j = 0; j < WIN; ++j) {
        const float4* k4 = (const float4*)(klat + (rowb + w[j]) * HDIM);
        float a = 0.f;
#pragma unroll
        for (int c = 0; c < 16; ++c) {
            float4 kv = k4[c];
            a += qr[c].x * kv.x + qr[c].y * kv.y + qr[c].z * kv.z + qr[c].w * kv.w;
        }
        sc[j] = a * 0.125f;   // 1/sqrt(64)
    }

    float mx = sc[0];
#pragma unroll
    for (int j = 1; j < WIN; ++j) mx = fmaxf(mx, sc[j]);
    float sum = 0.f;
#pragma unroll
    for (int j = 0; j < WIN; ++j) { sc[j] = __expf(sc[j] - mx); sum += sc[j]; }
    float inv = 1.f / sum;
#pragma unroll
    for (int j = 0; j < WIN; ++j) sc[j] *= inv;

    unsigned short* op = outbf + ((size_t)(b * SEQ + s)) * EMB + h * HDIM;
    const float* vbase = vlat + rowb * HDIM;
#pragma unroll
    for (int c = 0; c < 16; ++c) {
        float4 o = {0.f, 0.f, 0.f, 0.f};
#pragma unroll
        for (int j = 0; j < WIN; ++j) {
            float4 vv = ((const float4*)(vbase + (size_t)w[j] * HDIM))[c];
            o.x += sc[j] * vv.x; o.y += sc[j] * vv.y;
            o.z += sc[j] * vv.z; o.w += sc[j] * vv.w;
        }
        ushort4 ob;
        ob.x = f2bf(o.x); ob.y = f2bf(o.y); ob.z = f2bf(o.z); ob.w = f2bf(o.w);
        *(ushort4*)(op + c * 4) = ob;
    }
}

// ---------------------------------------------------------------------------
// Proj GEMM: out[M1,E] = attn_bf @ Wproj_bf^T + b_proj (f32 out -> d_out)
// ---------------------------------------------------------------------------
__global__ __launch_bounds__(256) void gemm_proj(const unsigned short* __restrict__ A,
                                                 const unsigned short* __restrict__ W,
                                                 const float* __restrict__ bias,
                                                 float* __restrict__ out) {
    int lane = threadIdx.x & 31, wid = threadIdx.x >> 5;
    int rowBase = blockIdx.y * 128 + (wid & 3) * 32;
    int colBase = blockIdx.x * 128 + (wid >> 2) * 64;
    v8f c[2][4] = {};
    gemm_mainloop<4>(A, EMB, W, EMB, rowBase, colBase, EMB, lane, c);

    int cl = lane & 15, rl = (lane & 16) ? 8 : 0;
#pragma unroll
    for (int mi = 0; mi < 2; ++mi)
#pragma unroll
        for (int ni = 0; ni < 4; ++ni) {
            int n = colBase + ni * 16 + cl;
            float bv = bias[n];
#pragma unroll
            for (int v = 0; v < 8; ++v) {
                int m = rowBase + mi * 16 + rl + v;
                out[(size_t)m * EMB + n] = c[mi][ni][v] + bv;
            }
        }
}

// ---------------------------------------------------------------------------
// Launch
// ---------------------------------------------------------------------------
extern "C" void kernel_launch(void* const* d_in, const int* in_sizes, int n_in,
                              void* d_out, int out_size, void* d_ws, size_t ws_size,
                              hipStream_t stream) {
    const float* x      = (const float*)d_in[0];
    const float* W_qkv  = (const float*)d_in[1];
    const float* b_qkv  = (const float*)d_in[2];
    const float* W_lat  = (const float*)d_in[3];
    const float* b_lat  = (const float*)d_in[4];
    const float* W_proj = (const float*)d_in[5];
    const float* b_proj = (const float*)d_in[6];

    // Workspace carve-up (256B aligned)
    size_t off = 0;
    auto carve = [&](size_t bytes) -> char* {
        char* p = (char*)d_ws + off;
        off += (bytes + 255) & ~(size_t)255;
        return p;
    };
    unsigned short* xbf    = (unsigned short*)carve((size_t)M1 * EMB * 2);   // 16 MB
    unsigned short* wqkvbf = (unsigned short*)carve((size_t)N1 * EMB * 2);   //  6 MB
    unsigned short* wprjbf = (unsigned short*)carve((size_t)EMB * EMB * 2);  //  2 MB
    unsigned short* wlatbf = (unsigned short*)carve((size_t)HDIM * HDIM * 2);
    float*          qf     = (float*)carve((size_t)M2 * HDIM * 4);           // 32 MB
    unsigned short* kbf    = (unsigned short*)carve((size_t)M2 * HDIM * 2);  // 16 MB
    unsigned short* vbf    = (unsigned short*)carve((size_t)M2 * HDIM * 2);  // 16 MB
    float*          klat   = (float*)carve((size_t)M2 * HDIM * 4);           // 32 MB
    float*          vlat   = (float*)carve((size_t)M2 * HDIM * 4);           // 32 MB
    unsigned short* attnbf = (unsigned short*)carve((size_t)M1 * EMB * 2);   // 16 MB

    // 1) convert inputs to bf16
    {
        int n;
        n = M1 * EMB;    cvt_f32_bf16<<<n / 4 / 256, 256, 0, stream>>>(x, xbf, n);
        n = N1 * EMB;    cvt_f32_bf16<<<n / 4 / 256, 256, 0, stream>>>(W_qkv, wqkvbf, n);
        n = EMB * EMB;   cvt_f32_bf16<<<n / 4 / 256, 256, 0, stream>>>(W_proj, wprjbf, n);
        n = HDIM * HDIM; cvt_f32_bf16<<<n / 4 / 256, 256, 0, stream>>>(W_lat, wlatbf, n);
    }

    // 2) QKV GEMM (8192 x 3072 x 1024)
    gemm_qkv<<<dim3(N1 / 128, M1 / 128), 256, 0, stream>>>(xbf, wqkvbf, b_qkv, qf, kbf, vbf);

    // 3) latent projections (131072 x 64 x 64) for k and v
    gemm_lat<<<dim3(1, M2 / 128), 256, 0, stream>>>(kbf, wlatbf, b_lat, klat);
    gemm_lat<<<dim3(1, M2 / 128), 256, 0, stream>>>(vbf, wlatbf, b_lat, vlat);

    // 4) windowed softmax attention
    attn_win<<<M2 / 256, 256, 0, stream>>>(qf, klat, vlat, attnbf);

    // 5) output projection (8192 x 1024 x 1024) -> d_out (f32)
    gemm_proj<<<dim3(EMB / 128, M1 / 128), 256, 0, stream>>>(attnbf, wprjbf, b_proj, (float*)d_out);
}
